// ProGenAttention_45964740001852
// MI455X (gfx1250) — matrix operand order repaired and checked by reading, not compiled, not run on hardware
//
#include <hip/hip_runtime.h>

// ---------------------------------------------------------------------------
// Types
// ---------------------------------------------------------------------------
typedef __attribute__((ext_vector_type(16))) __bf16         bf16x16;
typedef __attribute__((ext_vector_type(8)))  float          f32x8;
typedef __attribute__((ext_vector_type(4)))  float          f32x4;
typedef __attribute__((ext_vector_type(4)))  unsigned int   u32x4;
typedef __attribute__((ext_vector_type(4)))  unsigned short u16x4;

union FragBF {
    bf16x16 f;
    u32x4   h[2];   // two 16-byte halves
};

// fp32 -> bf16 (round-to-nearest-even), bit pattern only
__device__ __forceinline__ unsigned short f2bf(float x) {
    unsigned int u = __float_as_uint(x);
    unsigned int r = u + 0x7FFFu + ((u >> 16) & 1u);
    return (unsigned short)(r >> 16);
}

// generic (__shared__) pointer -> 32-bit LDS byte offset
__device__ __forceinline__ unsigned lds_off(const void* p) {
    return (unsigned)(unsigned long long)p;
}

// Problem constants
#define PB   2
#define PS   2048
#define PE   2048
#define PH   16
#define PDH  128
#define PROT 32
#define NQKV 6144
#define BSM  4096   // B*S

// ---------------------------------------------------------------------------
// Kernel: elementwise fp32 -> bf16 (vectorized x4)
// ---------------------------------------------------------------------------
__global__ __launch_bounds__(256)
void pg_f32_to_bf16(const float* __restrict__ src, unsigned short* __restrict__ dst,
                    int n4) {
    const int i = blockIdx.x * 256 + threadIdx.x;
    if (i < n4) {
        f32x4 v = reinterpret_cast<const f32x4*>(src)[i];
        u16x4 o = { f2bf(v.x), f2bf(v.y), f2bf(v.z), f2bf(v.w) };
        reinterpret_cast<u16x4*>(dst)[i] = o;
    }
}

// ---------------------------------------------------------------------------
// Kernel: C[M,N] = A[M,K] * W[N,K]^T   (bf16 in, f32 accumulate, fp32 out)
// Block: 128 threads = 4 waves (2x2), block tile 128x128, wave tile 64x64.
// Tiles staged to LDS with GLOBAL_LOAD_ASYNC_TO_LDS_B128, double-buffered.
// ---------------------------------------------------------------------------
#define BK 32
#define LROW 40   // padded LDS row (80 B): keeps 16B alignment, skews banks

__global__ __launch_bounds__(128)
void pg_gemm_bf16(const unsigned short* __restrict__ A,
                  const unsigned short* __restrict__ W,
                  float* __restrict__ C, int M, int N, int K) {
    __shared__ unsigned short As[2][128][LROW];
    __shared__ unsigned short Bs[2][128][LROW];

    const int tid  = threadIdx.x;
    const int lane = tid & 31;
    const int wave = tid >> 5;
    const int wm   = wave >> 1;     // 0..1
    const int wn   = wave & 1;      // 0..1
    const int m0   = blockIdx.y * 128;
    const int n0   = blockIdx.x * 128;
    const int mrow = lane & 15;
    const int hiA  = (lane >> 4) << 3;   // A-frag K base: 0 or 8 (elements)
    const int hiB  = (lane >> 4) << 3;   // B-frag K base: 0 or 8 (elements)

    // Per-thread staging: thread tid owns row tid of both tiles (64 B data).
    const unsigned ldsA = lds_off(&As[0][tid][0]);
    const unsigned ldsB = lds_off(&Bs[0][tid][0]);
    const unsigned bufStride = (unsigned)(128 * LROW * 2);   // bytes per buffer
    const unsigned rowA0 = (unsigned)(((size_t)(m0 + tid) * K) * 2);  // byte offset
    const unsigned rowB0 = (unsigned)(((size_t)(n0 + tid) * K) * 2);

    // issue one 128x32 bf16 tile of A and W into LDS buffer `buf` (async)
    auto issue_tile = [&](int buf, int k0) {
        const unsigned la = ldsA + (unsigned)buf * bufStride;
        const unsigned lb = ldsB + (unsigned)buf * bufStride;
        const unsigned ga = rowA0 + (unsigned)(k0 * 2);
        const unsigned gb = rowB0 + (unsigned)(k0 * 2);
        asm volatile("global_load_async_to_lds_b128 %0, %1, %2"
                     :: "v"(la), "v"(ga), "s"(A) : "memory");
        asm volatile("global_load_async_to_lds_b128 %0, %1, %2 offset:16"
                     :: "v"(la), "v"(ga), "s"(A) : "memory");
        asm volatile("global_load_async_to_lds_b128 %0, %1, %2 offset:32"
                     :: "v"(la), "v"(ga), "s"(A) : "memory");
        asm volatile("global_load_async_to_lds_b128 %0, %1, %2 offset:48"
                     :: "v"(la), "v"(ga), "s"(A) : "memory");
        asm volatile("global_load_async_to_lds_b128 %0, %1, %2"
                     :: "v"(lb), "v"(gb), "s"(W) : "memory");
        asm volatile("global_load_async_to_lds_b128 %0, %1, %2 offset:16"
                     :: "v"(lb), "v"(gb), "s"(W) : "memory");
        asm volatile("global_load_async_to_lds_b128 %0, %1, %2 offset:32"
                     :: "v"(lb), "v"(gb), "s"(W) : "memory");
        asm volatile("global_load_async_to_lds_b128 %0, %1, %2 offset:48"
                     :: "v"(lb), "v"(gb), "s"(W) : "memory");
    };

    f32x8 acc[4][4];
#pragma unroll
    for (int i = 0; i < 4; ++i)
#pragma unroll
        for (int j = 0; j < 4; ++j)
#pragma unroll
            for (int e = 0; e < 8; ++e) acc[i][j][e] = 0.0f;

    // software pipeline: prologue copy, then overlap copy(k+1) with wmma(k)
    issue_tile(0, 0);
    int cur = 0;
    for (int k0 = 0; k0 < K; k0 += BK) {
        asm volatile("s_wait_asynccnt 0x0" ::: "memory");
        __syncthreads();                       // tile `cur` resident for all waves
        if (k0 + BK < K) issue_tile(cur ^ 1, k0 + BK);

        FragBF af[4], bf[4];
#pragma unroll
        for (int i = 0; i < 4; ++i) {
            const int r = wm * 64 + i * 16 + mrow;
            af[i].h[0] = *reinterpret_cast<const u32x4*>(&As[cur][r][hiA]);
            af[i].h[1] = *reinterpret_cast<const u32x4*>(&As[cur][r][hiA + 16]);
            const int c = wn * 64 + i * 16 + mrow;
            bf[i].h[0] = *reinterpret_cast<const u32x4*>(&Bs[cur][c][hiB]);
            bf[i].h[1] = *reinterpret_cast<const u32x4*>(&Bs[cur][c][hiB + 16]);
        }
#pragma unroll
        for (int i = 0; i < 4; ++i)
#pragma unroll
            for (int j = 0; j < 4; ++j)
                acc[i][j] = __builtin_amdgcn_wmma_f32_16x16x32_bf16(
                    false, af[i].f, false, bf[j].f, (short)0, acc[i][j], false, false);
        cur ^= 1;
    }

    // ---- store: C layout row = r + (lane>>4)*8, col = lane&15 ----
#pragma unroll
    for (int i = 0; i < 4; ++i)
#pragma unroll
        for (int j = 0; j < 4; ++j)
#pragma unroll
            for (int r = 0; r < 8; ++r) {
                const int row = m0 + wm * 64 + i * 16 + r + ((lane >> 4) << 3);
                const int col = n0 + wn * 64 + j * 16 + (lane & 15);
                C[(size_t)row * N + col] = acc[i][j][r];
            }
}

// ---------------------------------------------------------------------------
// Kernel: RoPE + split qkv -> q,k bf16 [B,H,S,DH]; v bf16 transposed [B,H,DH,S]
// ---------------------------------------------------------------------------
__global__ __launch_bounds__(256)
void pg_rope_split(const float* __restrict__ qkv,
                   unsigned short* __restrict__ Qb,
                   unsigned short* __restrict__ Kb,
                   unsigned short* __restrict__ Vt) {
    const unsigned idx = blockIdx.x * 256u + threadIdx.x;   // 2^22 total
    const int d2 = idx & 63;
    const int h  = (idx >> 6) & 15;
    const int s  = (idx >> 10) & 2047;
    const int b  = idx >> 21;
    const int d0 = d2 * 2;
    const int d1 = d0 + 1;

    // map (h, d) -> qkv feature index (reference reshape through MP=8 groups)
    const int idx_h = h * PDH + d0;
    const int mp = idx_h >> 8;
    const int rr = idx_h & 255;
    const size_t base = ((size_t)(b * PS + s)) * NQKV + (size_t)mp * 768 + rr;

    float q0 = qkv[base],        q1 = qkv[base + 1];
    float k0 = qkv[base + 256],  k1 = qkv[base + 257];
    float v0 = qkv[base + 512],  v1 = qkv[base + 513];

    if (d0 < PROT) {
        const float inv = __powf(10000.0f, -(float)d0 / (float)PROT);
        const float ang = (float)s * inv;
        const float sn = __sinf(ang), cs = __cosf(ang);
        float nq0 = q0 * cs - q1 * sn;
        float nq1 = q1 * cs + q0 * sn;
        float nk0 = k0 * cs - k1 * sn;
        float nk1 = k1 * cs + k0 * sn;
        q0 = nq0; q1 = nq1; k0 = nk0; k1 = nk1;
    }

    const int bh = b * PH + h;
    const size_t qk_base = ((size_t)bh * PS + s) * PDH;
    Qb[qk_base + d0] = f2bf(q0);
    Qb[qk_base + d1] = f2bf(q1);
    Kb[qk_base + d0] = f2bf(k0);
    Kb[qk_base + d1] = f2bf(k1);
    Vt[((size_t)bh * PDH + d0) * PS + s] = f2bf(v0);
    Vt[((size_t)bh * PDH + d1) * PS + s] = f2bf(v1);
}

// ---------------------------------------------------------------------------
// Kernel: causal flash attention -> bf16 output in [B,S,E] layout.
// grid = (S/64, B*H); block = 128 threads = 4 waves; wave handles 16 queries.
// ---------------------------------------------------------------------------
__global__ __launch_bounds__(128)
void pg_flash_attn(const unsigned short* __restrict__ Q,
                   const unsigned short* __restrict__ Kb,
                   const unsigned short* __restrict__ Vt,
                   unsigned short* __restrict__ O) {
    __shared__ unsigned short Pl[4][16][LROW];   // per-wave P tile, padded rows

    const int tid  = threadIdx.x;
    const int lane = tid & 31;
    const int w    = tid >> 5;
    const int bh   = blockIdx.y;
    const int b    = bh >> 4;
    const int h    = bh & 15;
    const int qw0  = blockIdx.x * 64 + w * 16;
    const int mrow = lane & 15;
    const int hi8  = (lane >> 4) << 3;   // A-frag K base / C row base
    const int hi16 = (lane >> 4) << 3;   // B-frag K base
    const float scale = 0.08838834764831845f;  // 1/sqrt(128)

    // preload Q fragments (16 queries x DH=128 -> 4 bf16 A-frags)
    FragBF qf[4];
    {
        const unsigned short* qbase = Q + ((size_t)bh * PS + (qw0 + mrow)) * PDH;
#pragma unroll
        for (int ks = 0; ks < 4; ++ks) {
            qf[ks].h[0] = *reinterpret_cast<const u32x4*>(qbase + ks * 32 + hi8);
            qf[ks].h[1] = *reinterpret_cast<const u32x4*>(qbase + ks * 32 + hi8 + 16);
        }
    }

    float mrun[8], lrun[8];
    f32x8 o[8];
#pragma unroll
    for (int r = 0; r < 8; ++r) { mrun[r] = -3.0e38f; lrun[r] = 0.0f; }
#pragma unroll
    for (int dt = 0; dt < 8; ++dt)
#pragma unroll
        for (int e = 0; e < 8; ++e) o[dt][e] = 0.0f;

    for (int kv0 = 0; kv0 < qw0 + 16; kv0 += 32) {
        // ---- scores for 32 keys: two 16x16 tiles, 4 WMMAs each over DH ----
        f32x8 sc[2];
#pragma unroll
        for (int t = 0; t < 2; ++t) {
#pragma unroll
            for (int e = 0; e < 8; ++e) sc[t][e] = 0.0f;
            const unsigned short* kbase =
                Kb + ((size_t)bh * PS + (kv0 + t * 16 + mrow)) * PDH;
#pragma unroll
            for (int ks = 0; ks < 4; ++ks) {
                FragBF kf;
                kf.h[0] = *reinterpret_cast<const u32x4*>(kbase + ks * 32 + hi16);
                kf.h[1] = *reinterpret_cast<const u32x4*>(kbase + ks * 32 + hi16 + 16);
                sc[t] = __builtin_amdgcn_wmma_f32_16x16x32_bf16(
                    false, qf[ks].f, false, kf.f, (short)0, sc[t], false, false);
            }
        }

        // ---- causal mask + online softmax (row = 16-lane group reduction) ----
        float corr[8];
#pragma unroll
        for (int r = 0; r < 8; ++r) {
            const int qi  = qw0 + r + hi8;
            const int k0i = kv0 + mrow;
            const int k1i = kv0 + 16 + mrow;
            float v0 = sc[0][r] * scale; if (k0i > qi) v0 = -1.0e9f;
            float v1 = sc[1][r] * scale; if (k1i > qi) v1 = -1.0e9f;
            float mt = fmaxf(v0, v1);
#pragma unroll
            for (int off = 8; off > 0; off >>= 1)
                mt = fmaxf(mt, __shfl_xor(mt, off, 16));
            const float mn = fmaxf(mrun[r], mt);
            const float p0 = __expf(v0 - mn);
            const float p1 = __expf(v1 - mn);
            float rs = p0 + p1;
#pragma unroll
            for (int off = 8; off > 0; off >>= 1)
                rs += __shfl_xor(rs, off, 16);
            const float c = __expf(mrun[r] - mn);
            lrun[r] = lrun[r] * c + rs;
            mrun[r] = mn;
            corr[r] = c;
            Pl[w][r + hi8][mrow]      = f2bf(p0);
            Pl[w][r + hi8][16 + mrow] = f2bf(p1);
        }
#pragma unroll
        for (int dt = 0; dt < 8; ++dt)
#pragma unroll
            for (int r = 0; r < 8; ++r) o[dt][r] *= corr[r];

        // ---- P (C layout) -> A fragment via per-wave LDS round trip ----
        FragBF pf;
        pf.h[0] = *reinterpret_cast<const u32x4*>(&Pl[w][mrow][hi8]);
        pf.h[1] = *reinterpret_cast<const u32x4*>(&Pl[w][mrow][hi8 + 16]);

        // ---- O += P * V  (8 dh tiles; V transposed so keys contiguous) ----
#pragma unroll
        for (int dt = 0; dt < 8; ++dt) {
            FragBF vf;
            const unsigned short* vb =
                Vt + ((size_t)bh * PDH + dt * 16 + mrow) * PS + kv0;
            vf.h[0] = *reinterpret_cast<const u32x4*>(vb + hi16);
            vf.h[1] = *reinterpret_cast<const u32x4*>(vb + hi16 + 16);
            o[dt] = __builtin_amdgcn_wmma_f32_16x16x32_bf16(
                false, pf.f, false, vf.f, (short)0, o[dt], false, false);
        }
    }

    // ---- normalize + store bf16 to [B,S,E] ----
#pragma unroll
    for (int dt = 0; dt < 8; ++dt)
#pragma unroll
        for (int r = 0; r < 8; ++r) {
            const int qi = qw0 + r + hi8;
            const int dh = dt * 16 + mrow;
            O[((size_t)b * PS + qi) * PE + h * PDH + dh] = f2bf(o[dt][r] / lrun[r]);
        }
}

// ---------------------------------------------------------------------------
// Launcher
// ---------------------------------------------------------------------------
extern "C" void kernel_launch(void* const* d_in, const int* in_sizes, int n_in,
                              void* d_out, int out_size, void* d_ws, size_t ws_size,
                              hipStream_t stream) {
    const float* hidden = (const float*)d_in[0];
    const float* W_qkv  = (const float*)d_in[1];
    const float* W_out  = (const float*)d_in[2];
    float* out = (float*)d_out;

    char* ws = (char*)d_ws;
    const size_t qkvBytes   = (size_t)BSM * NQKV * sizeof(float);                  // ~100.7 MB
    const size_t hidBfBytes = (size_t)BSM * PE * 2;                                // 16.8 MB
    const size_t wqkBfBytes = (size_t)NQKV * PE * 2;                               // 25.2 MB
    const size_t wouBfBytes = (size_t)PE * PE * 2;                                 // 8.4 MB
    const size_t bf16Bytes  = (size_t)PB * PH * PS * PDH * 2;                      // 16.8 MB

    size_t off = 0;
    float*          qkv    = (float*)(ws + off);          off += qkvBytes;
    unsigned short* hidBf  = (unsigned short*)(ws + off); off += hidBfBytes;
    unsigned short* wqkBf  = (unsigned short*)(ws + off); off += wqkBfBytes;
    unsigned short* wouBf  = (unsigned short*)(ws + off); off += wouBfBytes;
    unsigned short* Qb     = (unsigned short*)(ws + off); off += bf16Bytes;
    unsigned short* Kb     = (unsigned short*)(ws + off); off += bf16Bytes;
    unsigned short* Vt     = (unsigned short*)(ws + off); off += bf16Bytes;
    unsigned short* attnBf = (unsigned short*)qkv;   // reuse qkv region after rope

    // 0) one-time fp32 -> bf16 conversions
    pg_f32_to_bf16<<<(BSM * PE / 4) / 256, 256, 0, stream>>>(hidden, hidBf, BSM * PE / 4);
    pg_f32_to_bf16<<<(NQKV * PE / 4) / 256, 256, 0, stream>>>(W_qkv, wqkBf, NQKV * PE / 4);
    pg_f32_to_bf16<<<(PE * PE / 4) / 256, 256, 0, stream>>>(W_out, wouBf, PE * PE / 4);

    // 1) QKV projection: [4096,2048] x [6144,2048]^T -> fp32 [4096,6144]
    pg_gemm_bf16<<<dim3(NQKV / 128, BSM / 128), 128, 0, stream>>>(
        hidBf, wqkBf, qkv, BSM, NQKV, PE);

    // 2) RoPE + split/transpose -> bf16 q,k,vT
    pg_rope_split<<<(PB * PS * PH * 64) / 256, 256, 0, stream>>>(qkv, Qb, Kb, Vt);

    // 3) causal flash attention -> bf16 [B,S,E]
    pg_flash_attn<<<dim3(PS / 64, PB * PH), 128, 0, stream>>>(Qb, Kb, Vt, attnBf);

    // 4) output projection: [4096,2048] x [2048,2048]^T -> d_out (fp32)
    pg_gemm_bf16<<<dim3(PE / 128, BSM / 128), 128, 0, stream>>>(
        attnBf, wouBf, out, BSM, PE, PE);
}
